// MRConv4d_35493609734667
// MI455X (gfx1250) — matrix operand order repaired and checked by reading, not compiled, not run on hardware
//
#include <hip/hip_runtime.h>
#include <hip/hip_fp16.h>
#include <hip/hip_bf16.h>

typedef _Float16 v8h  __attribute__((ext_vector_type(8)));
typedef _Float16 v16h __attribute__((ext_vector_type(16)));
typedef float    v8f  __attribute__((ext_vector_type(8)));

#define BB   32
#define CC   192
#define CO   384
#define HH   56
#define WWD  56
#define HW   3136
#define NPIX (BB * HW)      // 100352
#define KDIM 384
#define BN_EPS 1e-5f

// ---------------- BN fold: alpha = gamma*rsqrt(var+eps); betaE = beta + (b_conv - mean)*alpha
__global__ void bn_fold_kernel(const float* __restrict__ b_conv,
                               const float* __restrict__ gamma,
                               const float* __restrict__ beta,
                               const float* __restrict__ rmean,
                               const float* __restrict__ rvar,
                               float* __restrict__ alpha,
                               float* __restrict__ betaE) {
    int t = blockIdx.x * blockDim.x + threadIdx.x;
    if (t < CO) {
        float s = gamma[t] * rsqrtf(rvar[t] + BN_EPS);
        alpha[t] = s;
        betaE[t] = beta[t] + (b_conv[t] - rmean[t]) * s;
    }
}

// ---------------- Weight hi/lo f16 split (row-major [O][K], K contiguous)
__global__ void wsplit_kernel(const float* __restrict__ Wc,
                              _Float16* __restrict__ WH,
                              _Float16* __restrict__ WL) {
    int i = blockIdx.x * blockDim.x + threadIdx.x;
    if (i < CO * KDIM) {
        float v = Wc[i];
        _Float16 h = (_Float16)v;
        WH[i] = h;
        WL[i] = (_Float16)(v - (float)h);
    }
}

// ---------------- Per-(b,c) plane: max-relative aggregation via parity-class mins,
// write xc = [x | xj] into [pixel][K] hi/lo f16 arrays.
__global__ void __launch_bounds__(256)
prep_plane_kernel(const float* __restrict__ x,
                  _Float16* __restrict__ xcH,
                  _Float16* __restrict__ xcL) {
    __shared__ float plane[HW];
    __shared__ float cmin1[2][WWD], cmin2[2][WWD];
    __shared__ int   cidx [2][WWD];
    __shared__ float rmin1[2][HH],  rmin2[2][HH];
    __shared__ int   ridx [2][HH];

    const int bc = blockIdx.x;               // b*CC + c
    const float* xp = x + (size_t)bc * HW;
    for (int i = threadIdx.x; i < HW; i += blockDim.x) plane[i] = xp[i];
    __syncthreads();

    const int t = threadIdx.x;
    if (t < 112) {                            // per-column, per-row-parity mins
        int w = t % WWD, pr = t / WWD;
        float m1 = 1e30f, m2 = 1e30f; int id = -1;
        for (int h = pr; h < HH; h += 2) {
            float v = plane[h * WWD + w];
            if (v < m1) { m2 = m1; m1 = v; id = h; }
            else if (v < m2) { m2 = v; }
        }
        cmin1[pr][w] = m1; cmin2[pr][w] = m2; cidx[pr][w] = id;
    } else if (t < 224) {                     // per-row, per-col-parity mins
        int h = (t - 112) % HH, pr = (t - 112) / HH;
        float m1 = 1e30f, m2 = 1e30f; int id = -1;
        for (int w = pr; w < WWD; w += 2) {
            float v = plane[h * WWD + w];
            if (v < m1) { m2 = m1; m1 = v; id = w; }
            else if (v < m2) { m2 = v; }
        }
        rmin1[pr][h] = m1; rmin2[pr][h] = m2; ridx[pr][h] = id;
    }
    __syncthreads();

    const int b = bc / CC, c = bc % CC;
    for (int i = threadIdx.x; i < HW; i += blockDim.x) {
        int h = i / WWD, w = i - h * WWD;
        float v = plane[i];
        float cm = (cidx[h & 1][w] == h) ? cmin2[h & 1][w] : cmin1[h & 1][w];
        float rm = (ridx[w & 1][h] == w) ? rmin2[w & 1][h] : rmin1[w & 1][h];
        float mn = cm < rm ? cm : rm;
        float xj = v - mn; if (xj < 0.f) xj = 0.f;

        size_t base = ((size_t)(b * HW + i)) * KDIM;
        _Float16 h0 = (_Float16)v;
        xcH[base + c]      = h0;
        xcL[base + c]      = (_Float16)(v - (float)h0);
        _Float16 h1 = (_Float16)xj;
        xcH[base + CC + c] = h1;
        xcL[base + CC + c] = (_Float16)(xj - (float)h1);
    }
}

// ---------------- Fused WMMA GEMM + BN + exact GELU.
// A = W (M = out channel), B = xc (N = pixel). Per workgroup: 64 pixels x 128 o.
// B tile staged once per WG into LDS via global_load_async_to_lds_b128 (double
// buffered, ASYNCcnt), shared by all 8 waves; each wave keeps 4 accumulators
// (16 o x 64 pixels) and reuses its A fragment from registers.
__global__ void __launch_bounds__(256)
gemm_fused_kernel(const _Float16* __restrict__ xcH, const _Float16* __restrict__ xcL,
                  const _Float16* __restrict__ WH,  const _Float16* __restrict__ WL,
                  const float* __restrict__ alpha,  const float* __restrict__ betaE,
                  float* __restrict__ out) {
    // [buf][hi/lo][64 rows][32 halves] = 16 KB; only shared var -> LDS offset 0
    __shared__ _Float16 smem[2 * 2 * 64 * 32];

    // Capture the LDS buffer's address so alias analysis knows the inline-asm
    // async copies below can write it (otherwise the ds_loads fold to undef).
    asm volatile("" :: "v"(smem) : "memory");

    const int t    = threadIdx.x;
    const int wave = t >> 5;
    const int lane = t & 31;
    const int m    = lane & 15;      // A row (o) / B col (pixel) within tile
    const int hs   = lane >> 4;      // lane half

    const int pbase = blockIdx.x * 64;                 // 64-pixel block
    const int obase = (blockIdx.y * 8 + wave) * 16;    // 16-out-channel tile

    // --- async staging mapping: 256 threads = 64 rows x 4 x 16B segments
    const int srow = t >> 2;
    const int sseg = t & 3;
    const _Float16* gH = xcH + (size_t)(pbase + srow) * KDIM + sseg * 8;
    const _Float16* gL = xcL + (size_t)(pbase + srow) * KDIM + sseg * 8;
    const unsigned ldsRowHalves = (unsigned)(srow * 32 + sseg * 8);

    // --- A (weights), contiguous k order: lane half selects K 0..15 / 16..31 (matches B)
    const _Float16* wrH = WH + (size_t)(obase + m) * KDIM + hs * 16;
    const _Float16* wrL = WL + (size_t)(obase + m) * KDIM + hs * 16;

    v8f acc0 = {}, acc1 = {}, acc2 = {}, acc3 = {};

    auto stage = [&](int k, int buf) {
        unsigned offH = ((unsigned)(buf * 2 + 0) * 64u * 32u + ldsRowHalves) * 2u; // bytes
        unsigned offL = ((unsigned)(buf * 2 + 1) * 64u * 32u + ldsRowHalves) * 2u;
        // %2: LDS buffer pointer escapes into the asm -> its writes are visible.
        asm volatile("global_load_async_to_lds_b128 %0, %1, off"
                     :: "v"(offH), "v"(gH + k), "v"(smem) : "memory");
        asm volatile("global_load_async_to_lds_b128 %0, %1, off"
                     :: "v"(offL), "v"(gL + k), "v"(smem) : "memory");
    };

    stage(0, 0);
    asm volatile("s_wait_asynccnt 0x0" ::: "memory");
    __syncthreads();

    for (int kc = 0; kc < KDIM / 32; ++kc) {
        const int k   = kc * 32;
        const int buf = kc & 1;
        if (kc + 1 < KDIM / 32) stage(k + 32, buf ^ 1);  // prefetch next chunk

        v8h a0 = *(const v8h*)(wrH + k);
        v8h a1 = *(const v8h*)(wrH + k + 8);
        v8h a2 = *(const v8h*)(wrL + k);
        v8h a3 = *(const v8h*)(wrL + k + 8);
        v16h aH, aL;
        #pragma unroll
        for (int i = 0; i < 8; ++i) {
            aH[i] = a0[i]; aH[i + 8] = a1[i];
            aL[i] = a2[i]; aL[i + 8] = a3[i];
        }

        const _Float16* sH = smem + (buf * 2 + 0) * 64 * 32;
        const _Float16* sL = smem + (buf * 2 + 1) * 64 * 32;
        #pragma unroll
        for (int tn = 0; tn < 4; ++tn) {
            // B k order used here: lane = column, lane half selects K 0..15 / 16..31 (same as A)
            int roff = (tn * 16 + m) * 32 + hs * 16;
            v8h b0 = *(const v8h*)(sH + roff);
            v8h b1 = *(const v8h*)(sH + roff + 8);
            v8h b2 = *(const v8h*)(sL + roff);
            v8h b3 = *(const v8h*)(sL + roff + 8);
            v16h bH, bL;
            #pragma unroll
            for (int i = 0; i < 8; ++i) {
                bH[i] = b0[i]; bH[i + 8] = b1[i];
                bL[i] = b2[i]; bL[i + 8] = b3[i];
            }
            v8f a_ = (tn == 0) ? acc0 : (tn == 1) ? acc1 : (tn == 2) ? acc2 : acc3;
            // fp32-accurate via 2-term f16 split: hi*hi + hi*lo + lo*hi
            a_ = __builtin_amdgcn_wmma_f32_16x16x32_f16(false, aH, false, bH, (short)0, a_, false, false);
            a_ = __builtin_amdgcn_wmma_f32_16x16x32_f16(false, aH, false, bL, (short)0, a_, false, false);
            a_ = __builtin_amdgcn_wmma_f32_16x16x32_f16(false, aL, false, bH, (short)0, a_, false, false);
            if (tn == 0) acc0 = a_; else if (tn == 1) acc1 = a_; else if (tn == 2) acc2 = a_; else acc3 = a_;
        }

        asm volatile("s_wait_asynccnt 0x0" ::: "memory");
        __syncthreads();
    }

    // ---- epilogue: BN + exact GELU, pixel-contiguous stores
    float al[8], be[8];
    #pragma unroll
    for (int r = 0; r < 8; ++r) {
        int o = obase + r + hs * 8;   // D layout: VGPR r -> M = r + 8*hs
        al[r] = alpha[o];
        be[r] = betaE[o];
    }
    v8f accs[4] = {acc0, acc1, acc2, acc3};
    #pragma unroll
    for (int tn = 0; tn < 4; ++tn) {
        int pix = pbase + tn * 16 + m;        // N = lane&15
        int b   = pix / HW;
        int hw  = pix - b * HW;
        float* op = out + (size_t)b * CO * HW + hw;
        #pragma unroll
        for (int r = 0; r < 8; ++r) {
            int o = obase + r + hs * 8;
            float y = accs[tn][r] * al[r] + be[r];
            float g = 0.5f * y * (1.0f + erff(y * 0.70710678118654752f));
            op[(size_t)o * HW] = g;
        }
    }
}

extern "C" void kernel_launch(void* const* d_in, const int* in_sizes, int n_in,
                              void* d_out, int out_size, void* d_ws, size_t ws_size,
                              hipStream_t stream) {
    const float* x      = (const float*)d_in[0];
    const float* Wc     = (const float*)d_in[1];
    const float* b_conv = (const float*)d_in[2];
    const float* gamma  = (const float*)d_in[3];
    const float* beta   = (const float*)d_in[4];
    const float* rmean  = (const float*)d_in[5];
    const float* rvar   = (const float*)d_in[6];
    float* out = (float*)d_out;

    char* ws = (char*)d_ws;
    size_t off = 0;
    _Float16* xcH = (_Float16*)(ws + off); off += (size_t)NPIX * KDIM * sizeof(_Float16);
    _Float16* xcL = (_Float16*)(ws + off); off += (size_t)NPIX * KDIM * sizeof(_Float16);
    _Float16* WH  = (_Float16*)(ws + off); off += (size_t)CO * KDIM * sizeof(_Float16);
    _Float16* WL  = (_Float16*)(ws + off); off += (size_t)CO * KDIM * sizeof(_Float16);
    float* alpha  = (float*)(ws + off);    off += CO * sizeof(float);
    float* betaE  = (float*)(ws + off);    off += CO * sizeof(float);

    bn_fold_kernel<<<(CO + 255) / 256, 256, 0, stream>>>(b_conv, gamma, beta, rmean, rvar, alpha, betaE);
    wsplit_kernel<<<(CO * KDIM + 255) / 256, 256, 0, stream>>>(Wc, WH, WL);
    prep_plane_kernel<<<BB * CC, 256, 0, stream>>>(x, xcH, xcL);

    dim3 grid(NPIX / 64, CO / (16 * 8));   // 1568 pixel blocks x 3 (8 waves -> 128 o each)
    gemm_fused_kernel<<<grid, 256, 0, stream>>>(xcH, xcL, WH, WL, alpha, betaE, out);
}
